// MultiHeadAttention_65901978190342
// MI455X (gfx1250) — matrix operand order repaired and checked by reading, not compiled, not run on hardware
//
#include <hip/hip_runtime.h>

// ---------------------------------------------------------------------------
// MHA: B=1, S=4096, D=512, H=8, DEPTH=64.  fp32 in/out, bf16 WMMA compute.
// d_out = [out (4096*512 f32)] ++ [scores (8*4096*4096 f32)]
// ---------------------------------------------------------------------------

#define S_LEN 4096
#define D_DIM 512
#define H_NUM 8
#define DEPTH 64

typedef __attribute__((ext_vector_type(16))) __bf16 v16bf;
typedef __attribute__((ext_vector_type(8)))  float  v8f;

union Frag16 {
  v16bf v;
  unsigned short h[16];
  unsigned int   u[8];
  uint4          q[2];
};

__device__ __forceinline__ unsigned short f2bf(float f) {
  unsigned int u = __float_as_uint(f);
  return (unsigned short)((u + 0x7FFFu + ((u >> 16) & 1u)) >> 16);  // RNE
}

// ---- A-fragment loaders (16-bit A 16x32 layout; lanes0-15: K 0..7 & 16..23,
// ---- lanes16-31: K 8..15 & 24..31; 2 K-values per VGPR) ----
__device__ __forceinline__ v16bf load_a(const float* A, int lda, int r0, int k0, int lane) {
  const int r  = r0 + (lane & 15);
  const int kb = (lane & 16) ? 8 : 0;
  const float* p = A + (size_t)r * lda + k0 + kb;
  float4 x0 = *(const float4*)(p);
  float4 x1 = *(const float4*)(p + 4);
  float4 y0 = *(const float4*)(p + 16);
  float4 y1 = *(const float4*)(p + 20);
  Frag16 f;
  f.u[0] = (unsigned)f2bf(x0.x) | ((unsigned)f2bf(x0.y) << 16);
  f.u[1] = (unsigned)f2bf(x0.z) | ((unsigned)f2bf(x0.w) << 16);
  f.u[2] = (unsigned)f2bf(x1.x) | ((unsigned)f2bf(x1.y) << 16);
  f.u[3] = (unsigned)f2bf(x1.z) | ((unsigned)f2bf(x1.w) << 16);
  f.u[4] = (unsigned)f2bf(y0.x) | ((unsigned)f2bf(y0.y) << 16);
  f.u[5] = (unsigned)f2bf(y0.z) | ((unsigned)f2bf(y0.w) << 16);
  f.u[6] = (unsigned)f2bf(y1.x) | ((unsigned)f2bf(y1.y) << 16);
  f.u[7] = (unsigned)f2bf(y1.z) | ((unsigned)f2bf(y1.w) << 16);
  return f.v;
}

__device__ __forceinline__ v16bf load_a(const unsigned short* A, int lda, int r0, int k0, int lane) {
  const int r  = r0 + (lane & 15);
  const int kb = (lane & 16) ? 8 : 0;
  const unsigned short* p = A + (size_t)r * lda + k0 + kb;
  Frag16 f;
  f.q[0] = *(const uint4*)(p);        // K = kb .. kb+7
  f.q[1] = *(const uint4*)(p + 16);   // K = kb+16 .. kb+23
  return f.v;
}

// ---- B-fragment from LDS transposed tile bT[n][k] (ushort bT[64][32]) ----
// lanes 0-15: K 0..7 & 16..23, lanes 16-31: K 8..15 & 24..31, col n = lane%16 (+16*f4)
__device__ __forceinline__ v16bf load_b_lds(const unsigned short* bT, int f4, int lane) {
  const int n  = f4 * 16 + (lane & 15);
  const int kb = (lane & 16) ? 8 : 0;
  const unsigned short* p = bT + n * 32 + kb;
  Frag16 f;
  f.q[0] = *(const uint4*)(p);        // K = kb .. kb+7
  f.q[1] = *(const uint4*)(p + 16);   // K = kb+16 .. kb+23
  return f.v;
}

// ---- B-fragment where B[k][n] = Kmat[n][k] (rows of K are contiguous) ----
__device__ __forceinline__ v16bf load_b_rows(const unsigned short* Kmat, int ldb,
                                             int j0, int k0, int lane) {
  const int j  = j0 + (lane & 15);
  const int kb = (lane & 16) ? 8 : 0;
  const unsigned short* p = Kmat + (size_t)j * ldb + k0 + kb;
  Frag16 f;
  f.q[0] = *(const uint4*)(p);
  f.q[1] = *(const uint4*)(p + 16);
  return f.v;
}

// ---- stage B tile (32 k x 64 n) into LDS transposed [n][k], bf16 ----
__device__ __forceinline__ void stage_b(const float* B, int ldb, int k0, int n0,
                                        int tid, unsigned short* bT) {
#pragma unroll
  for (int s = 0; s < 2; ++s) {
    const int k = (tid >> 4) + s * 16;      // 0..31
    const int n = (tid & 15) * 4;           // 0..60
    const float4 x = *(const float4*)(B + (size_t)(k0 + k) * ldb + n0 + n);
    bT[(n + 0) * 32 + k] = f2bf(x.x);
    bT[(n + 1) * 32 + k] = f2bf(x.y);
    bT[(n + 2) * 32 + k] = f2bf(x.z);
    bT[(n + 3) * 32 + k] = f2bf(x.w);
  }
}

__device__ __forceinline__ void stage_b(const unsigned short* B, int ldb, int k0, int n0,
                                        int tid, unsigned short* bT) {
  const int k = tid >> 3;                   // 0..31
  const int n = (tid & 7) * 8;              // 0..56
  union { uint4 q; unsigned short h[8]; } x;
  x.q = *(const uint4*)(B + (size_t)(k0 + k) * ldb + n0 + n);
#pragma unroll
  for (int j = 0; j < 8; ++j) bT[(n + j) * 32 + k] = x.h[j];
}

__device__ __forceinline__ void store_c(float* p, float v)          { *p = v; }
__device__ __forceinline__ void store_c(unsigned short* p, float v) { *p = f2bf(v); }

// ---------------------------------------------------------------------------
// Generic bf16-WMMA GEMM: C[M x (64*gridDim.x)] = A[M x K] * B[K x N] (+bias)
// Block: 256 thr = 8 waves, tile 128(M) x 64(N); wave tile 16 x 64.
// Double-buffered LDS B staging (one barrier per k-step) + A-stream prefetch.
// CAUSAL: limit K-loop to this row-block's extent (A's upper cols are zero).
// ---------------------------------------------------------------------------
template <typename TA, typename TB, typename TO, bool HAS_BIAS, bool CAUSAL>
__global__ __launch_bounds__(256) void gemm_wmma(
    const TA* __restrict__ A, int lda, long long abstride,
    const TB* __restrict__ B, int ldb, int bcoloff,
    const float* __restrict__ bias,
    TO* __restrict__ C, int ldc, int ccoloff,
    int K) {
  __shared__ __align__(16) unsigned short bT[2][64 * 32];

  const int tid  = threadIdx.x;
  const int wave = tid >> 5;
  const int lane = tid & 31;
  const int m0   = blockIdx.y * 128 + wave * 16;
  const int n0   = blockIdx.x * 64;

  A += (long long)blockIdx.z * abstride;
  B += (long long)blockIdx.z * bcoloff;
  C += (long long)blockIdx.z * ccoloff;

  v8f acc[4];
#pragma unroll
  for (int f = 0; f < 4; ++f)
#pragma unroll
    for (int i = 0; i < 8; ++i) acc[f][i] = 0.0f;

  int kend = K;
  if (CAUSAL) {
    const int lim = ((int)blockIdx.y + 1) * 128;
    kend = lim < K ? lim : K;
  }

  // per-lane base row pointer of the A stream (for prefetch)
  const TA* arow = A + (size_t)(m0 + (lane & 15)) * lda;

  stage_b(B, ldb, 0, n0, tid, bT[0]);
  __syncthreads();

  int cur = 0;
  for (int k0 = 0; k0 < kend; k0 += 32) {
    if (k0 + 32 < kend) {
      stage_b(B, ldb, k0 + 32, n0, tid, bT[cur ^ 1]);   // fill next buffer
    }
    if (k0 + 64 < kend) {
      __builtin_prefetch(arow + k0 + 64, 0, 0);          // global_prefetch_b8
    }
    v16bf a = load_a(A, lda, m0, k0, lane);
#pragma unroll
    for (int f = 0; f < 4; ++f) {
      v16bf b = load_b_lds(bT[cur], f, lane);
      acc[f] = __builtin_amdgcn_wmma_f32_16x16x32_bf16(
          false, a, false, b, (short)0, acc[f], false, false);
    }
    __syncthreads();
    cur ^= 1;
  }

  const int ln = lane & 15;
  const int mo = (lane & 16) ? 8 : 0;
#pragma unroll
  for (int f = 0; f < 4; ++f) {
    const int n = n0 + f * 16 + ln;
    float bv = 0.0f;
    if (HAS_BIAS) bv = bias[n];
#pragma unroll
    for (int v = 0; v < 8; ++v) {
      const int m = m0 + mo + v;
      store_c(&C[(size_t)m * ldc + n], acc[f][v] + bv);
    }
  }
}

// ---------------------------------------------------------------------------
// scores_raw[h,i,j] = (Q[i,h,:] . K[j,h,:]) / DEPTH, masked (j>i) -> -1e9
// Wave tiles entirely above the diagonal skip the matmul.
// ---------------------------------------------------------------------------
__global__ __launch_bounds__(256) void qkt_scores(
    const unsigned short* __restrict__ Qb,
    const unsigned short* __restrict__ Kb,
    float* __restrict__ scores) {
  const int tid  = threadIdx.x;
  const int wave = tid >> 5;
  const int lane = tid & 31;
  const int h    = blockIdx.z;
  const int m0   = blockIdx.y * 128 + wave * 16;  // query rows i
  const int n0   = blockIdx.x * 64;               // key cols j
  const int ln   = lane & 15;
  const int mo   = (lane & 16) ? 8 : 0;

  float* sp = scores + (size_t)h * S_LEN * S_LEN;

  if (n0 > m0 + 15) {  // whole wave-tile masked: j_min > i_max
#pragma unroll
    for (int f = 0; f < 4; ++f)
#pragma unroll
      for (int v = 0; v < 8; ++v)
        sp[(size_t)(m0 + mo + v) * S_LEN + n0 + f * 16 + ln] = -1e9f;
    return;
  }

  v8f acc[4];
#pragma unroll
  for (int f = 0; f < 4; ++f)
#pragma unroll
    for (int i = 0; i < 8; ++i) acc[f][i] = 0.0f;

#pragma unroll
  for (int k0 = 0; k0 < DEPTH; k0 += 32) {
    v16bf a = load_a(Qb, D_DIM, m0, h * DEPTH + k0, lane);
#pragma unroll
    for (int f = 0; f < 4; ++f) {
      v16bf b = load_b_rows(Kb, D_DIM, n0 + f * 16, h * DEPTH + k0, lane);
      acc[f] = __builtin_amdgcn_wmma_f32_16x16x32_bf16(
          false, a, false, b, (short)0, acc[f], false, false);
    }
  }

  const float scale = 1.0f / (float)DEPTH;  // reference divides by dk, not sqrt
#pragma unroll
  for (int f = 0; f < 4; ++f) {
#pragma unroll
    for (int v = 0; v < 8; ++v) {
      const int i = m0 + mo + v;
      const int j = n0 + f * 16 + ln;
      float raw = acc[f][v] * scale;
      if (j > i) raw = -1e9f;
      sp[(size_t)i * S_LEN + j] = raw;
    }
  }
}

// ---------------------------------------------------------------------------
// In-place row softmax over 4096 cols. One block per (h, i) row.
// Masked entries (-1e9) underflow exp() to exactly 0, matching reference.
// ---------------------------------------------------------------------------
__global__ __launch_bounds__(256) void softmax_rows(float* __restrict__ scores) {
  __shared__ float red[256];
  const int t = threadIdx.x;
  float* row = scores + ((size_t)blockIdx.y * S_LEN + blockIdx.x) * S_LEN;

  float r[16];
  float lm = -3.4e38f;
#pragma unroll
  for (int e = 0; e < 16; ++e) {
    r[e] = row[t + e * 256];
    lm = fmaxf(lm, r[e]);
  }
  red[t] = lm;
  __syncthreads();
  for (int s = 128; s > 0; s >>= 1) {
    if (t < s) red[t] = fmaxf(red[t], red[t + s]);
    __syncthreads();
  }
  const float mx = red[0];
  __syncthreads();

  float ls = 0.0f;
#pragma unroll
  for (int e = 0; e < 16; ++e) {
    r[e] = __expf(r[e] - mx);
    ls += r[e];
  }
  red[t] = ls;
  __syncthreads();
  for (int s = 128; s > 0; s >>= 1) {
    if (t < s) red[t] += red[t + s];
    __syncthreads();
  }
  const float inv = 1.0f / red[0];
#pragma unroll
  for (int e = 0; e < 16; ++e) row[t + e * 256] = r[e] * inv;
}

// ---------------------------------------------------------------------------
extern "C" void kernel_launch(void* const* d_in, const int* in_sizes, int n_in,
                              void* d_out, int out_size, void* d_ws, size_t ws_size,
                              hipStream_t stream) {
  (void)in_sizes; (void)n_in; (void)out_size; (void)ws_size;

  const float* values  = (const float*)d_in[0];
  const float* keys    = (const float*)d_in[1];
  const float* queries = (const float*)d_in[2];
  /* d_in[3] = mask: causal, computed analytically instead of loading 64MB */
  const float* Wq = (const float*)d_in[4];
  const float* bq = (const float*)d_in[5];
  const float* Wk = (const float*)d_in[6];
  const float* bk = (const float*)d_in[7];
  const float* Wv = (const float*)d_in[8];
  const float* bv = (const float*)d_in[9];
  const float* Wo = (const float*)d_in[10];
  const float* bo = (const float*)d_in[11];

  float* out    = (float*)d_out;                      // [4096, 512]
  float* scores = out + (size_t)S_LEN * D_DIM;        // [8, 4096, 4096]

  unsigned short* Qbf  = (unsigned short*)d_ws;       // bf16 [4096,512]
  unsigned short* Kbf  = Qbf + (size_t)S_LEN * D_DIM;
  unsigned short* Vbf  = Kbf + (size_t)S_LEN * D_DIM;
  unsigned short* Attn = Vbf + (size_t)S_LEN * D_DIM; // bf16 [4096,512]

  const dim3 blk(256);
  const dim3 gProj(D_DIM / 64, S_LEN / 128, 1);       // (8, 32)

  // Q/K/V projections: f32 in -> bf16 out, f32 accum, +bias
  gemm_wmma<float, float, unsigned short, true, false>
      <<<gProj, blk, 0, stream>>>(queries, D_DIM, 0, Wq, D_DIM, 0, bq,
                                  Qbf, D_DIM, 0, D_DIM);
  gemm_wmma<float, float, unsigned short, true, false>
      <<<gProj, blk, 0, stream>>>(keys, D_DIM, 0, Wk, D_DIM, 0, bk,
                                  Kbf, D_DIM, 0, D_DIM);
  gemm_wmma<float, float, unsigned short, true, false>
      <<<gProj, blk, 0, stream>>>(values, D_DIM, 0, Wv, D_DIM, 0, bv,
                                  Vbf, D_DIM, 0, D_DIM);

  // raw scores = Q K^T / depth, causal mask
  qkt_scores<<<dim3(S_LEN / 64, S_LEN / 128, H_NUM), blk, 0, stream>>>(
      Qbf, Kbf, scores);

  // in-place softmax over each of the 8*4096 rows
  softmax_rows<<<dim3(S_LEN, H_NUM, 1), blk, 0, stream>>>(scores);

  // attn = P @ V   (A = f32 scores from d_out, converted in-reg; causal k-limit)
  gemm_wmma<float, unsigned short, unsigned short, false, true>
      <<<dim3(1, S_LEN / 128, H_NUM), blk, 0, stream>>>(
          scores, S_LEN, (long long)S_LEN * S_LEN,
          Vbf, D_DIM, DEPTH, nullptr,
          Attn, D_DIM, DEPTH, S_LEN);

  // out = attn @ Wo + bo  (f32 out)
  gemm_wmma<unsigned short, float, float, true, false>
      <<<gProj, blk, 0, stream>>>(Attn, D_DIM, 0, Wo, D_DIM, 0, bo,
                                  out, D_DIM, 0, D_DIM);
}
